// GraphTransformer_29532195127715
// MI455X (gfx1250) — matrix operand order repaired and checked by reading, not compiled, not run on hardware
//
#include <hip/hip_runtime.h>
#include <hip/hip_bf16.h>

// ---------------------------------------------------------------------------
// CDNA5 / gfx1250 graph-transformer. wave32. WMMA bf16 (f32 accum) for all
// GEMMs; async global->LDS weight staging; atomics for segment ops.
// Workspace bump-allocated from d_ws.
// ---------------------------------------------------------------------------

typedef __attribute__((ext_vector_type(16))) __bf16 v16bf;
typedef __attribute__((ext_vector_type(8)))  float  v8f;

union AF { v16bf v; __bf16 e[16]; };
union CF { v8f  v; float  f[8];  };

__device__ __forceinline__ int f2ord(float f) {
  int i = __float_as_int(f);
  return i >= 0 ? i : (i ^ 0x7fffffff);
}
__device__ __forceinline__ float ord2f(int k) {
  return __int_as_float(k >= 0 ? k : (k ^ 0x7fffffff));
}

// ---------------------------------------------------------------------------
// Generic GEMM: C[M x NC] (+)= act(A[M x K] @ W[K x NC] + bias)
//
// Pipeline per block:
//   1) issue GLOBAL_LOAD_ASYNC_TO_LDS_B128 for the f32 weight panel (ASYNCcnt)
//   2) overlap: load A fragments (global b128) and convert to bf16 in VGPRs
//   3) s_wait_asynccnt 0 + barrier
//   4) pack LDS f32 weights -> fragment-major bf16 LDS (one aligned 32B
//      ds_load per lane per B fragment in the main loop)
//   5) WMMA main loop
//
// 4 waves / block, each wave owns a 16-row tile across all NC columns.
// CS: compile-time C row stride. ACT: 0 none, 1 leaky_relu(0.01).
// BETA: 0 store, 1 accumulate into C.
// ---------------------------------------------------------------------------
template<int K, int NC, int CS, int ACT, int BETA>
__global__ __launch_bounds__(128) void k_gemm(
    const float* __restrict__ A, const float* __restrict__ W,
    const float* __restrict__ bias, float* __restrict__ C, int M)
{
  constexpr int KC = K / 32;   // 32-deep K chunks
  constexpr int NT = NC / 16;  // 16-wide N tiles

  __shared__ __align__(16) float  Wstage[K * NC];   // raw f32, async-staged
  __shared__ __align__(32) __bf16 Wp[K * NC];       // fragment-major bf16

  // ---- 1) async-stage the weight panel into LDS (no VGPR results) ------
  {
    const uint32_t lds0 = (uint32_t)(uintptr_t)(&Wstage[0]);
    const int base = threadIdx.x * 4;               // 4 floats per b128
#pragma unroll
    for (int it = 0; it < (K * NC) / 512; ++it) {   // 512 floats per round
      const int idx = base + it * 512;
      const uint32_t laddr = lds0 + (uint32_t)(idx * 4);
      const float* g = W + idx;
      asm volatile("global_load_async_to_lds_b128 %0, %1, off"
                   :: "v"(laddr), "v"(g)
                   : "memory");
    }
  }

  const int wave = threadIdx.x >> 5;
  const int lane = threadIdx.x & 31;
  const int half = lane >> 4;
  const int lid  = lane & 15;
  const int rowTile = blockIdx.x * 64 + wave * 16;

  // ---- 2) A fragments (overlapped with async weight staging) -----------
  // ISA 16-bit A 16x32 layout: lanes0-15 K=[0..7|16..23], lanes16-31 K=[8..15|24..31]
  AF a[KC];
  const int arow = rowTile + lid;          // both lane halves carry same rows
  const bool inb = arow < M;
  const float* Ar = A + (size_t)arow * K;
#pragma unroll
  for (int kc = 0; kc < KC; ++kc) {
    float4 r0 = {0, 0, 0, 0}, r1 = r0, r2 = r0, r3 = r0;
    if (inb) {
      const float4* p0 = reinterpret_cast<const float4*>(Ar + kc * 32 + half * 8);
      const float4* p1 = reinterpret_cast<const float4*>(Ar + kc * 32 + half * 8 + 16);
      r0 = p0[0]; r1 = p0[1];
      r2 = p1[0]; r3 = p1[1];
    }
    a[kc].e[0]  = (__bf16)r0.x; a[kc].e[1]  = (__bf16)r0.y;
    a[kc].e[2]  = (__bf16)r0.z; a[kc].e[3]  = (__bf16)r0.w;
    a[kc].e[4]  = (__bf16)r1.x; a[kc].e[5]  = (__bf16)r1.y;
    a[kc].e[6]  = (__bf16)r1.z; a[kc].e[7]  = (__bf16)r1.w;
    a[kc].e[8]  = (__bf16)r2.x; a[kc].e[9]  = (__bf16)r2.y;
    a[kc].e[10] = (__bf16)r2.z; a[kc].e[11] = (__bf16)r2.w;
    a[kc].e[12] = (__bf16)r3.x; a[kc].e[13] = (__bf16)r3.y;
    a[kc].e[14] = (__bf16)r3.z; a[kc].e[15] = (__bf16)r3.w;
  }

  // ---- 3) wait for this wave's async loads; barrier covers all waves ---
  asm volatile("s_wait_asynccnt 0x0" ::: "memory");
  __syncthreads();

  // ---- 4) pack f32 LDS -> fragment-major bf16 LDS ----------------------
  // Packed layout: [kc][nt][lane][16 bf16]
  for (int p = threadIdx.x; p < K * NC; p += 128) {
    int e    = p & 15;
    int ln   = (p >> 4) & 31;
    int nt   = (p >> 9) % NT;
    int kc   = (p >> 9) / NT;
    int hf   = ln >> 4;
    int ld   = ln & 15;
    // B 32x16 layout: lanes0-15 col=lid K=[0..7|16..23], lanes16-31 col=lid K=[8..15|24..31]
    int kk  = kc * 32 + hf * 8 + (e & 7) + ((e >> 3) << 4);
    int col = nt * 16 + ld;
    Wp[p] = (__bf16)Wstage[kk * NC + col];
  }
  __syncthreads();

  // ---- 5) WMMA main loop ----------------------------------------------
  const int rbase = rowTile + half * 8;    // D rows this lane-half writes
#pragma unroll
  for (int nt = 0; nt < NT; ++nt) {
    CF acc;
#pragma unroll
    for (int i = 0; i < 8; ++i) acc.f[i] = 0.0f;
#pragma unroll
    for (int kc = 0; kc < KC; ++kc) {
      AF b;
      b.v = *reinterpret_cast<const v16bf*>(&Wp[((kc * NT + nt) * 32 + lane) * 16]);
      acc.v = __builtin_amdgcn_wmma_f32_16x16x32_bf16(
          false, a[kc].v, false, b.v, (short)0, acc.v, false, false);
    }
    const int col = nt * 16 + lid;
    const float bz = bias ? bias[col] : 0.0f;
    float* dp = C + (size_t)rbase * CS + col;
    if (rbase + 8 <= M) {
#pragma unroll
      for (int i = 0; i < 8; ++i) {
        float val = acc.f[i] + bz;
        if (ACT == 1) val = val > 0.0f ? val : 0.01f * val;
        if (BETA) dp[i * CS] += val; else dp[i * CS] = val;
      }
    } else {
#pragma unroll
      for (int i = 0; i < 8; ++i) {
        if (rbase + i < M) {
          float val = acc.f[i] + bz;
          if (ACT == 1) val = val > 0.0f ? val : 0.01f * val;
          if (BETA) dp[i * CS] += val; else dp[i * CS] = val;
        }
      }
    }
  }
}

// ---------------------------------------------------------------------------
// Fill kernels
// ---------------------------------------------------------------------------
__global__ void k_fill_f(float* p, long n, float v) {
  long i = (long)blockIdx.x * 256 + threadIdx.x;
  if (i < n) p[i] = v;
}
__global__ void k_fill_i(int* p, long n, int v) {
  long i = (long)blockIdx.x * 256 + threadIdx.x;
  if (i < n) p[i] = v;
}

// ---------------------------------------------------------------------------
// Graph construction
// ---------------------------------------------------------------------------
__global__ void k_build_edges(const int* __restrict__ ei, const int* __restrict__ batch,
                              int* __restrict__ src, int* __restrict__ dst,
                              int E, int N, int NAUG) {
  int i = blockIdx.x * 256 + threadIdx.x;
  int E2 = E + 2 * N, EA = E2 + NAUG;
  if (i >= EA) return;
  if (i < E)            { src[i] = ei[i];              dst[i] = ei[E + i]; }
  else if (i < E + N)   { int u = i - E;     src[i] = u;              dst[i] = batch[u] + N; }
  else if (i < E2)      { int u = i - E - N; src[i] = batch[u] + N;   dst[i] = u; }
  else                  { int j = i - E2;    src[i] = j;              dst[i] = j; }
}

__global__ void k_ebase(const float* __restrict__ ea, float* __restrict__ e_aug,
                        int E, long total) {
  long idx = (long)blockIdx.x * 256 + threadIdx.x;
  if (idx >= total) return;
  long e = idx >> 6; int d = (int)(idx & 63);
  e_aug[idx] = (e < (long)E) ? ea[idx] : (d == 0 ? 1.0f : 0.0f);
}

__global__ void k_deg(const int* __restrict__ dst, float* __restrict__ cnt, int E2) {
  int i = blockIdx.x * 256 + threadIdx.x;
  if (i < E2) atomicAdd(&cnt[dst[i]], 1.0f);
}

__global__ void k_loop_scatter(const float* __restrict__ e_aug, const int* __restrict__ dst,
                               float* __restrict__ acc, long total) {
  long idx = (long)blockIdx.x * 256 + threadIdx.x;
  if (idx >= total) return;
  long e = idx >> 6; int d = (int)(idx & 63);
  float v = e_aug[idx];
  if (v != 0.0f) atomicAdd(&acc[(size_t)dst[e] * 64 + d], v);
}

__global__ void k_loop_fin(const float* __restrict__ acc, const float* __restrict__ cnt,
                           float* __restrict__ e_aug, long base, long total) {
  long idx = (long)blockIdx.x * 256 + threadIdx.x;
  if (idx >= total) return;
  long i = idx >> 6;
  e_aug[base + idx] = acc[idx] / fmaxf(cnt[i], 1.0f);
}

__global__ void k_augbatch(const int* __restrict__ batch, int* __restrict__ ab,
                           float* __restrict__ cntn, int N, int NAUG) {
  int i = blockIdx.x * 256 + threadIdx.x;
  if (i >= NAUG) return;
  int g = (i < N) ? batch[i] : (i - N);
  ab[i] = g;
  if (i < N) atomicAdd(&cntn[g], 1.0f);
}

__global__ void k_init_h(const float* __restrict__ x, const float* __restrict__ cond,
                         float* __restrict__ h, int N, long total) {
  long idx = (long)blockIdx.x * 256 + threadIdx.x;
  if (idx >= total) return;
  long i = idx >> 6;
  h[idx] = (i < (long)N) ? x[idx] : cond[idx - (size_t)N * 64];
}

// ---------------------------------------------------------------------------
// Graph LayerNorm (scalar mean/var per graph over all elements)
// ---------------------------------------------------------------------------
__global__ __launch_bounds__(256) void k_ln_sum(const float* __restrict__ X,
                                                const int* __restrict__ ab,
                                                float* __restrict__ gsum, int n) {
  int i = blockIdx.x * 8 + (threadIdx.x >> 5);
  int lane = threadIdx.x & 31;
  if (i >= n) return;
  float t = X[(size_t)i * 64 + lane] + X[(size_t)i * 64 + 32 + lane];
  for (int m = 16; m > 0; m >>= 1) t += __shfl_xor(t, m, 32);
  if (lane == 0) atomicAdd(&gsum[ab[i]], t);
}
__global__ void k_ln_mean(float* __restrict__ gmean, const float* __restrict__ gsum,
                          const float* __restrict__ cntn, int G) {
  int g = blockIdx.x * 256 + threadIdx.x;
  if (g < G) gmean[g] = gsum[g] / ((cntn[g] + 1.0f) * 64.0f);
}
__global__ __launch_bounds__(256) void k_ln_var(const float* __restrict__ X,
                                                const int* __restrict__ ab,
                                                const float* __restrict__ gmean,
                                                float* __restrict__ gvar, int n) {
  int i = blockIdx.x * 8 + (threadIdx.x >> 5);
  int lane = threadIdx.x & 31;
  if (i >= n) return;
  float m = gmean[ab[i]];
  float d0 = X[(size_t)i * 64 + lane] - m;
  float d1 = X[(size_t)i * 64 + 32 + lane] - m;
  float t = d0 * d0 + d1 * d1;
  for (int mm = 16; mm > 0; mm >>= 1) t += __shfl_xor(t, mm, 32);
  if (lane == 0) atomicAdd(&gvar[ab[i]], t);
}
__global__ void k_ln_rstd(float* __restrict__ grstd, const float* __restrict__ gvar,
                          const float* __restrict__ cntn, int G) {
  int g = blockIdx.x * 256 + threadIdx.x;
  if (g < G) grstd[g] = rsqrtf(gvar[g] / ((cntn[g] + 1.0f) * 64.0f) + 1e-5f);
}
__global__ void k_ln_apply(const float* __restrict__ X, const int* __restrict__ ab,
                           const float* __restrict__ gmean, const float* __restrict__ grstd,
                           float* __restrict__ out, int ostride, long total) {
  long idx = (long)blockIdx.x * 256 + threadIdx.x;
  if (idx >= total) return;
  long i = idx >> 6; int d = (int)(idx & 63);
  int g = ab[i];
  out[(size_t)i * ostride + d] = (X[idx] - gmean[g]) * grstd[g];
}

// ---------------------------------------------------------------------------
// gen_conv: agg starts as hn (the "+x" term), then messages are scattered:
// agg[dst] += relu(hn[src] + e) + eps.  hn = cat[:, :64].
// ---------------------------------------------------------------------------
__global__ void k_init_agg(float* __restrict__ agg, const float* __restrict__ cat, long total) {
  long idx = (long)blockIdx.x * 256 + threadIdx.x;
  if (idx >= total) return;
  agg[idx] = cat[(idx >> 6) * 128 + (idx & 63)];
}

__global__ void k_gen_msg(const float* __restrict__ cat, const float* __restrict__ e_aug,
                          const int* __restrict__ src, const int* __restrict__ dst,
                          float* __restrict__ agg, long total) {
  long idx = (long)blockIdx.x * 256 + threadIdx.x;
  if (idx >= total) return;
  long e = idx >> 6; int d = (int)(idx & 63);
  float m = cat[(size_t)src[e] * 128 + d] + e_aug[idx];
  m = fmaxf(m, 0.0f) + 1e-7f;
  atomicAdd(&agg[(size_t)dst[e] * 64 + d], m);
}

// ---------------------------------------------------------------------------
// Attention: alpha = q[dst] . (k[src] + ee) / 8 ; wave per edge, shfl reduce.
// ---------------------------------------------------------------------------
__global__ __launch_bounds__(256) void k_alpha(const float* __restrict__ q,
                                               const float* __restrict__ kk,
                                               const float* __restrict__ ee,
                                               const int* __restrict__ src,
                                               const int* __restrict__ dst,
                                               float* __restrict__ alpha,
                                               int* __restrict__ mkey, int EA) {
  int e = blockIdx.x * 8 + (threadIdx.x >> 5);
  int lane = threadIdx.x & 31;
  if (e >= EA) return;
  int s_ = src[e], d_ = dst[e];
#pragma unroll
  for (int h = 0; h < 2; ++h) {
    int o = h * 64;
    float acc =
        q[(size_t)d_ * 128 + o + lane] *
            (kk[(size_t)s_ * 128 + o + lane] + ee[(size_t)e * 128 + o + lane]) +
        q[(size_t)d_ * 128 + o + 32 + lane] *
            (kk[(size_t)s_ * 128 + o + 32 + lane] + ee[(size_t)e * 128 + o + 32 + lane]);
    for (int m = 16; m > 0; m >>= 1) acc += __shfl_xor(acc, m, 32);
    if (lane == 0) {
      float al = acc * 0.125f;   // / sqrt(64)
      alpha[e * 2 + h] = al;
      atomicMax(&mkey[d_ * 2 + h], f2ord(al));
    }
  }
}

__global__ void k_exp(float* __restrict__ alpha, const int* __restrict__ dst,
                      const int* __restrict__ mkey, float* __restrict__ s, int EA) {
  int idx = blockIdx.x * 256 + threadIdx.x;
  if (idx >= EA * 2) return;
  int e = idx >> 1, h = idx & 1;
  int d_ = dst[e];
  float a = __expf(alpha[idx] - ord2f(mkey[d_ * 2 + h]));
  alpha[idx] = a;
  atomicAdd(&s[d_ * 2 + h], a);
}

__global__ void k_msg(const float* __restrict__ v, const float* __restrict__ ee,
                      const float* __restrict__ alpha, const float* __restrict__ s,
                      const int* __restrict__ src, const int* __restrict__ dst,
                      float* __restrict__ tout, long total) {
  long idx = (long)blockIdx.x * 256 + threadIdx.x;
  if (idx >= total) return;
  long e = idx >> 7; int d = (int)(idx & 127); int h = d >> 6;
  int s_ = src[e], d_ = dst[e];
  float a = alpha[e * 2 + h] / (s[d_ * 2 + h] + 1e-16f);
  float val = (v[(size_t)s_ * 128 + d] + ee[(size_t)e * 128 + d]) * a;
  atomicAdd(&tout[(size_t)d_ * 128 + d], val);
}

// ---------------------------------------------------------------------------
// Outputs
// ---------------------------------------------------------------------------
__global__ void k_copy(const float* __restrict__ a, float* __restrict__ b, long n) {
  long i = (long)blockIdx.x * 256 + threadIdx.x;
  if (i < n) b[i] = a[i];
}
__global__ void k_gacc(const float* __restrict__ h, const int* __restrict__ batch,
                       float* __restrict__ gacc, long total) {
  long idx = (long)blockIdx.x * 256 + threadIdx.x;
  if (idx >= total) return;
  atomicAdd(&gacc[(size_t)batch[idx >> 6] * 64 + (idx & 63)], h[idx]);
}
__global__ void k_glob(const float* __restrict__ gacc, const float* __restrict__ cntn,
                       const float* __restrict__ h, float* __restrict__ out,
                       int N, long total) {
  long idx = (long)blockIdx.x * 256 + threadIdx.x;
  if (idx >= total) return;
  long g = idx >> 6; int d = (int)(idx & 63);
  out[idx] = gacc[idx] / fmaxf(cntn[g], 1.0f) + h[(size_t)(N + g) * 64 + d];
}
__global__ void k_ne(const float* __restrict__ h, const int* __restrict__ nei,
                     float* __restrict__ out, int NE, long total) {
  long idx = (long)blockIdx.x * 256 + threadIdx.x;
  if (idx >= total) return;
  long e = idx >> 6; int d = (int)(idx & 63);
  out[idx] = h[(size_t)nei[e] * 64 + d] + h[(size_t)nei[NE + e] * 64 + d];
}

// ---------------------------------------------------------------------------
static inline int gdiv(long a, long b) { return (int)((a + b - 1) / b); }

extern "C" void kernel_launch(void* const* d_in, const int* in_sizes, int n_in,
                              void* d_out, int out_size, void* d_ws, size_t ws_size,
                              hipStream_t stream) {
  const float* x         = (const float*)d_in[0];
  const float* cond      = (const float*)d_in[1];
  const float* edge_attr = (const float*)d_in[2];
  const int*   ei        = (const int*)d_in[3];
  const int*   nei       = (const int*)d_in[4];
  const int*   batch     = (const int*)d_in[5];
  const float* gen_w  = (const float*)d_in[6];
  const float* gen_b  = (const float*)d_in[7];
  const float* q_w    = (const float*)d_in[8];
  const float* q_b    = (const float*)d_in[9];
  const float* k_w    = (const float*)d_in[10];
  const float* k_b    = (const float*)d_in[11];
  const float* v_w    = (const float*)d_in[12];
  const float* v_b    = (const float*)d_in[13];
  const float* e_w    = (const float*)d_in[14];
  const float* skip_w = (const float*)d_in[15];
  const float* skip_b = (const float*)d_in[16];
  const float* lin_w  = (const float*)d_in[17];
  const float* lin_b  = (const float*)d_in[18];
  const float* ff_w1  = (const float*)d_in[19];
  const float* ff_b1  = (const float*)d_in[20];
  const float* ff_w2  = (const float*)d_in[21];
  const float* ff_b2  = (const float*)d_in[22];

  const int N    = in_sizes[0] / 64;
  const int G    = in_sizes[1] / 64;
  const int E    = in_sizes[2] / 64;
  const int NE   = in_sizes[4] / 2;
  const int NAUG = N + G;
  const int E2   = E + 2 * N;
  const int EA   = E2 + NAUG;

  // ---- workspace bump allocation --------------------------------------
  char* wp = (char*)d_ws;
  auto alloc = [&](size_t bytes) -> void* {
    void* r = (void*)wp;
    wp += (bytes + 255) & ~(size_t)255;
    return r;
  };
  int*   srcA    = (int*)  alloc((size_t)EA * 4);
  int*   dstA    = (int*)  alloc((size_t)EA * 4);
  float* e_aug   = (float*)alloc((size_t)EA * 64 * 4);
  float* ee      = (float*)alloc((size_t)EA * 128 * 4);
  float* alpha   = (float*)alloc((size_t)EA * 2 * 4);
  float* h       = (float*)alloc((size_t)NAUG * 64 * 4);
  float* agg     = (float*)alloc((size_t)NAUG * 64 * 4);
  float* lh      = (float*)alloc((size_t)NAUG * 64 * 4);
  float* ln2     = (float*)alloc((size_t)NAUG * 64 * 4);
  float* loopacc = (float*)alloc((size_t)NAUG * 64 * 4);
  float* cat     = (float*)alloc((size_t)NAUG * 128 * 4);
  float* qb      = (float*)alloc((size_t)NAUG * 128 * 4);
  float* kb      = (float*)alloc((size_t)NAUG * 128 * 4);
  float* vb      = (float*)alloc((size_t)NAUG * 128 * 4);
  float* tout    = (float*)alloc((size_t)NAUG * 128 * 4);
  float* ffmid   = (float*)alloc((size_t)NAUG * 256 * 4);
  float* cnt     = (float*)alloc((size_t)NAUG * 4);
  float* cntn    = (float*)alloc((size_t)G * 4);
  float* gsum    = (float*)alloc((size_t)G * 4);
  float* gvar    = (float*)alloc((size_t)G * 4);
  float* gmean   = (float*)alloc((size_t)G * 4);
  float* grstd   = (float*)alloc((size_t)G * 4);
  int*   mkey    = (int*)  alloc((size_t)NAUG * 2 * 4);
  float* sden    = (float*)alloc((size_t)NAUG * 2 * 4);
  int*   ab      = (int*)  alloc((size_t)NAUG * 4);
  float* gacc    = (float*)alloc((size_t)G * 64 * 4);

  float* out0 = (float*)d_out;                 // n_emb  [N,64]
  float* out1 = out0 + (size_t)N * 64;         // glob   [G,64]
  float* out2 = out1 + (size_t)G * 64;         // ne_emb [NE,64]

  const long NA64  = (long)NAUG * 64;
  const long EA64  = (long)EA * 64;
  const long EA128 = (long)EA * 128;

  // ---- graph construction ---------------------------------------------
  k_fill_f<<<gdiv(NAUG, 256), 256, 0, stream>>>(cnt, NAUG, 0.0f);
  k_fill_f<<<gdiv(NA64, 256), 256, 0, stream>>>(loopacc, NA64, 0.0f);
  k_fill_f<<<gdiv(G, 256), 256, 0, stream>>>(cntn, G, 0.0f);
  k_build_edges<<<gdiv(EA, 256), 256, 0, stream>>>(ei, batch, srcA, dstA, E, N, NAUG);
  k_ebase<<<gdiv((long)E2 * 64, 256), 256, 0, stream>>>(edge_attr, e_aug, E, (long)E2 * 64);
  k_deg<<<gdiv(E2, 256), 256, 0, stream>>>(dstA, cnt, E2);
  k_loop_scatter<<<gdiv((long)E2 * 64, 256), 256, 0, stream>>>(e_aug, dstA, loopacc, (long)E2 * 64);
  k_loop_fin<<<gdiv(NA64, 256), 256, 0, stream>>>(loopacc, cnt, e_aug, (long)E2 * 64, NA64);
  k_augbatch<<<gdiv(NAUG, 256), 256, 0, stream>>>(batch, ab, cntn, N, NAUG);
  k_init_h<<<gdiv(NA64, 256), 256, 0, stream>>>(x, cond, h, N, NA64);

  const int gM   = gdiv(NAUG, 64);   // node-GEMM grid
  const int gE   = gdiv(EA, 64);     // edge-GEMM grid
  const int gWav = gdiv(NAUG, 8);    // wave-per-node

  for (int i = 0; i < 3; ++i) {
    // ---- LN1: hn -> cat[:, :64] -------------------------------------
    k_fill_f<<<gdiv(G, 256), 256, 0, stream>>>(gsum, G, 0.0f);
    k_fill_f<<<gdiv(G, 256), 256, 0, stream>>>(gvar, G, 0.0f);
    k_ln_sum<<<gWav, 256, 0, stream>>>(h, ab, gsum, NAUG);
    k_ln_mean<<<gdiv(G, 256), 256, 0, stream>>>(gmean, gsum, cntn, G);
    k_ln_var<<<gWav, 256, 0, stream>>>(h, ab, gmean, gvar, NAUG);
    k_ln_rstd<<<gdiv(G, 256), 256, 0, stream>>>(grstd, gvar, cntn, G);
    k_ln_apply<<<gdiv(NA64, 256), 256, 0, stream>>>(h, ab, gmean, grstd, cat, 128, NA64);

    // ---- gen_conv: cat[:, 64:] = (msg_agg + hn) @ gen_w + gen_b ------
    k_init_agg<<<gdiv(NA64, 256), 256, 0, stream>>>(agg, cat, NA64);
    k_gen_msg<<<gdiv(EA64, 256), 256, 0, stream>>>(cat, e_aug, srcA, dstA, agg, EA64);
    k_gemm<64, 64, 128, 0, 0><<<gM, 128, 0, stream>>>(
        agg, gen_w + (size_t)i * 64 * 64, gen_b + (size_t)i * 64, cat + 64, NAUG);

    // ---- trans_conv ---------------------------------------------------
    k_gemm<128, 128, 128, 0, 0><<<gM, 128, 0, stream>>>(
        cat, q_w + (size_t)i * 128 * 128, q_b + (size_t)i * 128, qb, NAUG);
    k_gemm<128, 128, 128, 0, 0><<<gM, 128, 0, stream>>>(
        cat, k_w + (size_t)i * 128 * 128, k_b + (size_t)i * 128, kb, NAUG);
    k_gemm<128, 128, 128, 0, 0><<<gM, 128, 0, stream>>>(
        cat, v_w + (size_t)i * 128 * 128, v_b + (size_t)i * 128, vb, NAUG);
    // ee = e_aug @ e_w  (the big one: 750k x 64 @ 64 x 128)
    k_gemm<64, 128, 128, 0, 0><<<gE, 128, 0, stream>>>(
        e_aug, e_w + (size_t)i * 64 * 128, nullptr, ee, EA);
    // softmax over incoming edges
    k_fill_i<<<gdiv((long)NAUG * 2, 256), 256, 0, stream>>>(mkey, (long)NAUG * 2, (int)0x80000000);
    k_alpha<<<gdiv(EA, 8), 256, 0, stream>>>(qb, kb, ee, srcA, dstA, alpha, mkey, EA);
    k_fill_f<<<gdiv((long)NAUG * 2, 256), 256, 0, stream>>>(sden, (long)NAUG * 2, 0.0f);
    k_exp<<<gdiv((long)EA * 2, 256), 256, 0, stream>>>(alpha, dstA, mkey, sden, EA);
    // tout = cat @ skip_w + skip_b, then scatter attention messages on top
    k_gemm<128, 128, 128, 0, 0><<<gM, 128, 0, stream>>>(
        cat, skip_w + (size_t)i * 128 * 128, skip_b + (size_t)i * 128, tout, NAUG);
    k_msg<<<gdiv(EA128, 256), 256, 0, stream>>>(vb, ee, alpha, sden, srcA, dstA, tout, EA128);
    // lh = tout @ lin_w + lin_b
    k_gemm<128, 64, 64, 0, 0><<<gM, 128, 0, stream>>>(
        tout, lin_w + (size_t)i * 128 * 64, lin_b + (size_t)i * 64, lh, NAUG);

    // ---- LN2 ----------------------------------------------------------
    k_fill_f<<<gdiv(G, 256), 256, 0, stream>>>(gsum, G, 0.0f);
    k_fill_f<<<gdiv(G, 256), 256, 0, stream>>>(gvar, G, 0.0f);
    k_ln_sum<<<gWav, 256, 0, stream>>>(lh, ab, gsum, NAUG);
    k_ln_mean<<<gdiv(G, 256), 256, 0, stream>>>(gmean, gsum, cntn, G);
    k_ln_var<<<gWav, 256, 0, stream>>>(lh, ab, gmean, gvar, NAUG);
    k_ln_rstd<<<gdiv(G, 256), 256, 0, stream>>>(grstd, gvar, cntn, G);
    k_ln_apply<<<gdiv(NA64, 256), 256, 0, stream>>>(lh, ab, gmean, grstd, ln2, 64, NA64);

    // ---- FFN with residual: h += W2( leaky(W1(ln2)) ) -----------------
    k_gemm<64, 256, 256, 1, 0><<<gM, 128, 0, stream>>>(
        ln2, ff_w1 + (size_t)i * 64 * 256, ff_b1 + (size_t)i * 256, ffmid, NAUG);
    k_gemm<256, 64, 64, 0, 1><<<gM, 128, 0, stream>>>(
        ffmid, ff_w2 + (size_t)i * 256 * 64, ff_b2 + (size_t)i * 64, h, NAUG);
  }

  // ---- outputs ---------------------------------------------------------
  k_copy<<<gdiv((long)N * 64, 256), 256, 0, stream>>>(h, out0, (long)N * 64);
  k_fill_f<<<gdiv((long)G * 64, 256), 256, 0, stream>>>(gacc, (long)G * 64, 0.0f);
  k_gacc<<<gdiv((long)N * 64, 256), 256, 0, stream>>>(h, batch, gacc, (long)N * 64);
  k_glob<<<gdiv((long)G * 64, 256), 256, 0, stream>>>(gacc, cntn, h, out1, N, (long)G * 64);
  k_ne<<<gdiv((long)NE * 64, 256), 256, 0, stream>>>(h, nei, out2, NE, (long)NE * 64);
}
